// LoraInjectedLinear_51857435132115
// MI455X (gfx1250) — compile-verified, matrix operand order checked by reading, not run on hardware
//
#include <hip/hip_runtime.h>
#include <hip/hip_bf16.h>
#include <stdint.h>

// ---------------------------------------------------------------------------
// LoRA-injected linear:  out = x @ (W + down@up)^T + bias
//   x: [8192, 4096] f32 (B*S flattened), W: [4096, 4096] f32,
//   down: [4096,16], up: [16,4096], bias: [4096], out: [8192,4096] f32
// Strategy: fold LoRA into W once, convert x/W to bf16 in workspace, then a
// double-buffered, async-to-LDS tiled GEMM on v_wmma_f32_16x16x32_bf16.
// Block tile 256x128 (8 waves, 4x2), wave tile 64x64 (4x4 WMMA accums).
// ---------------------------------------------------------------------------

typedef __attribute__((ext_vector_type(16))) __bf16 v16bf;
typedef __attribute__((ext_vector_type(8)))  float  v8f;
typedef __attribute__((ext_vector_type(4)))  int    v4i;

typedef __attribute__((address_space(1))) v4i* gptr_v4i;  // global int4*
typedef __attribute__((address_space(3))) v4i* lptr_v4i;  // LDS int4*

static constexpr int Mdim = 4 * 2048;   // B*S
static constexpr int Ndim = 4096;       // OUT
static constexpr int Kdim = 4096;       // IN
static constexpr int Rdim = 16;

static constexpr int BM = 256, BN = 128, BK = 64;
static constexpr int LDA = 72;          // padded row stride (bank rotation)
static constexpr int KSLABS = Kdim / BK;

#if __has_builtin(__builtin_amdgcn_global_load_async_to_lds_b128) && \
    __has_builtin(__builtin_amdgcn_s_wait_asynccnt)
#define USE_ASYNC 1
#else
#define USE_ASYNC 0
#endif

__device__ __forceinline__ uint16_t f2bf(float f) {
  union { float f; uint32_t u; } c; c.f = f;
  uint32_t u = c.u;
  uint32_t r = u + 0x7FFFu + ((u >> 16) & 1u);   // round-to-nearest-even
  if ((u & 0x7F800000u) == 0x7F800000u) r = u;   // Inf/NaN passthrough
  return (uint16_t)(r >> 16);
}

__device__ __forceinline__ uint4 pack8(const float* f) {
  uint32_t w0 = (uint32_t)f2bf(f[0]) | ((uint32_t)f2bf(f[1]) << 16);
  uint32_t w1 = (uint32_t)f2bf(f[2]) | ((uint32_t)f2bf(f[3]) << 16);
  uint32_t w2 = (uint32_t)f2bf(f[4]) | ((uint32_t)f2bf(f[5]) << 16);
  uint32_t w3 = (uint32_t)f2bf(f[6]) | ((uint32_t)f2bf(f[7]) << 16);
  return make_uint4(w0, w1, w2, w3);
}

// --- x (f32) -> bf16, 8 elements per thread ---------------------------------
__global__ __launch_bounds__(256)
void cvt_x_kernel(const float* __restrict__ x, uint16_t* __restrict__ xb, int n8) {
  int i = blockIdx.x * blockDim.x + threadIdx.x;
  if (i >= n8) return;
  const float4* p = reinterpret_cast<const float4*>(x) + (size_t)i * 2;
  union { float4 v[2]; float f[8]; } t;
  t.v[0] = p[0];
  t.v[1] = p[1];
  reinterpret_cast<uint4*>(xb)[i] = pack8(t.f);
}

// --- Wb = bf16(old_W + down @ up), 8 elements per thread --------------------
__global__ __launch_bounds__(256)
void prep_w_kernel(const float* __restrict__ w, const float* __restrict__ down,
                   const float* __restrict__ up, uint16_t* __restrict__ wb) {
  int i = blockIdx.x * blockDim.x + threadIdx.x;     // one per 8 elems
  if (i >= Ndim * Kdim / 8) return;
  int n  = i >> 9;                  // / (Kdim/8)
  int k8 = (i & 511) * 8;
  union { float4 v[2]; float f[8]; } acc;
  const float4* wp = reinterpret_cast<const float4*>(w + (size_t)n * Kdim + k8);
  acc.v[0] = wp[0];
  acc.v[1] = wp[1];
#pragma unroll
  for (int r = 0; r < Rdim; ++r) {
    float d = down[n * Rdim + r];
    const float4* upr = reinterpret_cast<const float4*>(up + (size_t)r * Kdim + k8);
    union { float4 v[2]; float f[8]; } u;
    u.v[0] = upr[0];
    u.v[1] = upr[1];
#pragma unroll
    for (int j = 0; j < 8; ++j) acc.f[j] = fmaf(d, u.f[j], acc.f[j]);
  }
  reinterpret_cast<uint4*>(wb)[i] = pack8(acc.f);
}

// --- async (or sync-fallback) 16B global -> LDS copy ------------------------
__device__ __forceinline__ void cp16(const uint16_t* __restrict__ g,
                                     uint16_t* __restrict__ l) {
#if USE_ASYNC
  // low 32 bits of a generic LDS address == LDS byte offset (ISA aperture rule)
  __builtin_amdgcn_global_load_async_to_lds_b128(
      (gptr_v4i)(uintptr_t)g,
      (lptr_v4i)(uint32_t)(uintptr_t)l,
      0, 0);
#else
  *reinterpret_cast<uint4*>(l) = *reinterpret_cast<const uint4*>(g);
#endif
}

__device__ __forceinline__ void wait_async_all() {
#if USE_ASYNC
  __builtin_amdgcn_s_wait_asynccnt(0);
#endif
}

// --- Main GEMM: C[M,N] = Xb[M,K] * Wb[N,K]^T + bias -------------------------
union FragU { v16bf v; uint4 q[2]; };

__global__ __launch_bounds__(256)
void lora_gemm_kernel(const uint16_t* __restrict__ Xb,
                      const uint16_t* __restrict__ Wb,
                      const float* __restrict__ bias,
                      float* __restrict__ out) {
  __shared__ __align__(16) uint16_t As[2][BM * LDA];   // 2 x 36 KiB
  __shared__ __align__(16) uint16_t Bs[2][BN * LDA];   // 2 x 18 KiB

  const int tid  = threadIdx.x;
  const int lane = tid & 31;
  const int wave = tid >> 5;
  const int wm   = (wave & 3) * 64;   // wave row offset in block tile (4 in M)
  const int wn   = (wave >> 2) * 64;  // wave col offset in block tile (2 in N)
  const int bM   = blockIdx.y * BM;
  const int bN   = blockIdx.x * BN;

  const int l15  = lane & 15;
  const int koff = (lane >> 4) * 8;   // K half selector per ISA A/B layout

  // global->LDS copy pattern: 8 threads per 64-wide row, b128 per thread
  const int lrow = tid >> 3;          // 0..31
  const int lcol = (tid & 7) * 8;     // 0..56

  v8f acc[4][4] = {};

  // ---- prefetch slab 0 into buffer 0 ----
#pragma unroll
  for (int p = 0; p < 8; ++p)
    cp16(Xb + (size_t)(bM + lrow + p * 32) * Kdim + lcol,
         &As[0][(lrow + p * 32) * LDA + lcol]);
#pragma unroll
  for (int p = 0; p < 4; ++p)
    cp16(Wb + (size_t)(bN + lrow + p * 32) * Kdim + lcol,
         &Bs[0][(lrow + p * 32) * LDA + lcol]);

  for (int it = 0; it < KSLABS; ++it) {
    const int cur = it & 1;
    wait_async_all();      // our slab-`it` copies have landed in LDS
    __syncthreads();       // everyone's copies landed; buf cur^1 fully consumed

    if (it + 1 < KSLABS) { // prefetch slab it+1 into the other buffer
      const int k0n = (it + 1) * BK;
#pragma unroll
      for (int p = 0; p < 8; ++p)
        cp16(Xb + (size_t)(bM + lrow + p * 32) * Kdim + k0n + lcol,
             &As[cur ^ 1][(lrow + p * 32) * LDA + lcol]);
#pragma unroll
      for (int p = 0; p < 4; ++p)
        cp16(Wb + (size_t)(bN + lrow + p * 32) * Kdim + k0n + lcol,
             &Bs[cur ^ 1][(lrow + p * 32) * LDA + lcol]);
    }

    // ---- compute on buffer `cur` ----
#pragma unroll
    for (int kk = 0; kk < BK; kk += 32) {
      FragU a[4], b[4];
#pragma unroll
      for (int mi = 0; mi < 4; ++mi) {
        const uint16_t* p0 = &As[cur][(wm + mi * 16 + l15) * LDA + kk + koff];
        a[mi].q[0] = *reinterpret_cast<const uint4*>(p0);        // K koff..+7
        a[mi].q[1] = *reinterpret_cast<const uint4*>(p0 + 16);   // K koff+16..+23
      }
#pragma unroll
      for (int ni = 0; ni < 4; ++ni) {
        const uint16_t* p0 = &Bs[cur][(wn + ni * 16 + l15) * LDA + kk + koff];
        b[ni].q[0] = *reinterpret_cast<const uint4*>(p0);
        b[ni].q[1] = *reinterpret_cast<const uint4*>(p0 + 16);
      }
#pragma unroll
      for (int mi = 0; mi < 4; ++mi)
#pragma unroll
        for (int ni = 0; ni < 4; ++ni)
          acc[mi][ni] = __builtin_amdgcn_wmma_f32_16x16x32_bf16(
              false, a[mi].v, false, b[ni].v, (short)0, acc[mi][ni], false, false);
    }
  }

  // Epilogue: C layout -> lane<16: elem r = row r; lane>=16: elem r = row r+8
#pragma unroll
  for (int ni = 0; ni < 4; ++ni) {
    int gc = bN + wn + ni * 16 + l15;
    float bv = bias[gc];
#pragma unroll
    for (int mi = 0; mi < 4; ++mi) {
      int rbase = bM + wm + mi * 16 + (lane >> 4) * 8;
#pragma unroll
      for (int r = 0; r < 8; ++r)
        out[(size_t)(rbase + r) * Ndim + gc] = acc[mi][ni][r] + bv;
    }
  }
}

extern "C" void kernel_launch(void* const* d_in, const int* in_sizes, int n_in,
                              void* d_out, int out_size, void* d_ws, size_t ws_size,
                              hipStream_t stream) {
  const float* x      = (const float*)d_in[0];
  const float* old_w  = (const float*)d_in[1];
  const float* old_b  = (const float*)d_in[2];
  const float* ldown  = (const float*)d_in[3];
  const float* lup    = (const float*)d_in[4];
  float* out = (float*)d_out;

  uint16_t* Xb = (uint16_t*)d_ws;                       // 8192*4096*2 = 64 MiB
  uint16_t* Wb = Xb + (size_t)Mdim * Kdim;              // + 32 MiB

  int nx8 = Mdim * Kdim / 8;
  cvt_x_kernel<<<(nx8 + 255) / 256, 256, 0, stream>>>(x, Xb, nx8);

  int nw8 = Ndim * Kdim / 8;
  prep_w_kernel<<<(nw8 + 255) / 256, 256, 0, stream>>>(old_w, ldown, lup, Wb);

  dim3 grid(Ndim / BN, Mdim / BM);
  lora_gemm_kernel<<<grid, 256, 0, stream>>>(Xb, Wb, old_b, out);
}
